// ContrastiveCorrelationLoss_29068338660027
// MI455X (gfx1250) — matrix operand order repaired and checked by reading, not compiled, not run on hardware
//
#include <hip/hip_runtime.h>
#include <hip/hip_bf16.h>
#include <math.h>

// ---- problem constants (from reference) ----
#define N_   16
#define CF_  768
#define CC_  64
#define H_   56
#define W_   56
#define S_   40
#define P_   (S_*S_)          // 1600 sample points per image
#define PP_  1664             // P_ padded to a multiple of 128 (zero rows)
#define KT_  (CF_/32)         // 24 K-steps for the feature GEMM
#define LDST 40               // LDS row stride in halfs (80B: 16B aligned, bank-conflict free)

typedef __attribute__((ext_vector_type(16))) _Float16 v16h;
typedef __attribute__((ext_vector_type(8)))  _Float16 v8h;
typedef __attribute__((ext_vector_type(8)))  float    v8f;
typedef int i32x4v __attribute__((__vector_size__(16)));
using g_i32x4 = __attribute__((address_space(1))) i32x4v;   // global int4
using l_i32x4 = __attribute__((address_space(3))) i32x4v;   // LDS int4

// ---------------- fragment loaders ----------------
// A (16x32 f16, M x K): lane (g = lane>>4, m = lane&15) holds row m,
// K = g*8..g*8+7 and K = 16+g*8..+7 (two 16B chunks from a K-contiguous row)
__device__ __forceinline__ v16h load_a_frag(const _Float16* p) {
    v8h lo = *reinterpret_cast<const v8h*>(p);
    v8h hi = *reinterpret_cast<const v8h*>(p + 16);
    return __builtin_shufflevector(lo, hi, 0,1,2,3,4,5,6,7,8,9,10,11,12,13,14,15);
}
// B (32x16 f16, K x N) stored column-major: lane (g, n) holds column n,
// K = g*8..g*8+7 and K = 16+g*8..+7 (two 16B chunks)
__device__ __forceinline__ v16h load_b_frag(const _Float16* p) {
    v8h lo = *reinterpret_cast<const v8h*>(p);
    v8h hi = *reinterpret_cast<const v8h*>(p + 16);
    return __builtin_shufflevector(lo, hi, 0,1,2,3,4,5,6,7,8,9,10,11,12,13,14,15);
}

__device__ __forceinline__ v8f wmma_f16(v16h a, v16h b, v8f c) {
    return __builtin_amdgcn_wmma_f32_16x16x32_f16(
        false, a, false, b, (short)0, c, false, false);
}

// ---------------- async global->LDS copy (CDNA5 path, guarded) ----------------
__device__ __forceinline__ void cp_async16(_Float16* ldsDst, const _Float16* gSrc) {
#if __has_builtin(__builtin_amdgcn_global_load_async_to_lds_b128)
    __builtin_amdgcn_global_load_async_to_lds_b128(
        (g_i32x4*)(uintptr_t)gSrc,          // addrspace(1) int4*
        (l_i32x4*)(uintptr_t)ldsDst,        // addrspace(3) int4* (low 32 bits = LDS offset)
        0, 0);
#else
    *reinterpret_cast<uint4*>(ldsDst) = *reinterpret_cast<const uint4*>(gSrc);
#endif
}
template <int NV>
__device__ __forceinline__ void cp_wait_le() {
#if __has_builtin(__builtin_amdgcn_s_wait_asynccnt)
    __builtin_amdgcn_s_wait_asynccnt(NV);   // immediate operand required
#endif
}

// ---------------- kernel 0a: zero accumulators ----------------
__global__ void k_zero(float* __restrict__ buf, int n) {
    int i = blockIdx.x * blockDim.x + threadIdx.x;
    if (i < n) buf[i] = 0.0f;
}
// ---------------- kernel 0b: zero pad rows (rows 1600..1663 per image) ----------------
__global__ void k_zero_pad(_Float16* __restrict__ buf, int cols, int total16) {
    int i = blockIdx.x * blockDim.x + threadIdx.x;   // index of 8-half chunks
    if (i >= total16) return;
    const int per = (PP_ - P_) * cols / 8;           // chunks per image
    const int n = i / per, r = i - n * per;
    uint4 z; z.x = z.y = z.z = z.w = 0u;
    reinterpret_cast<uint4*>(buf + ((size_t)n * PP_ + P_) * cols)[r] = z;
}

// ---------------- kernel 1: bilinear sample + L2 normalize -> f16 ----------------
__global__ __launch_bounds__(256) void k_sample(
    const float* __restrict__ feats, const float* __restrict__ code,
    const float* __restrict__ coords,
    _Float16* __restrict__ outF, _Float16* __restrict__ outC)
{
    const int p = blockIdx.x;            // 0..1599 (= h*S + w)
    const int n = blockIdx.y;            // 0..15
    const int h = p / S_, w = p % S_;
    // g[n,h,w,:] = coords[n,w,h,:]*2-1 ; pixel = (g+1)/2*(dim-1) = coord*(dim-1)
    const float cx = coords[(((size_t)n * S_ + w) * S_ + h) * 2 + 0];
    const float cy = coords[(((size_t)n * S_ + w) * S_ + h) * 2 + 1];
    float x = fminf(fmaxf(cx * (float)(W_ - 1), 0.0f), (float)(W_ - 1));
    float y = fminf(fmaxf(cy * (float)(H_ - 1), 0.0f), (float)(H_ - 1));
    int x0 = (int)floorf(x), y0 = (int)floorf(y);
    float wx = x - (float)x0, wy = y - (float)y0;
    int x1 = min(x0 + 1, W_ - 1), y1 = min(y0 + 1, H_ - 1);
    const float w00 = (1.f - wx) * (1.f - wy), w01 = wx * (1.f - wy);
    const float w10 = (1.f - wx) * wy,        w11 = wx * wy;
    const int i00 = y0 * W_ + x0, i01 = y0 * W_ + x1;
    const int i10 = y1 * W_ + x0, i11 = y1 * W_ + x1;

    const int t = threadIdx.x;
    float vf[3];
    float ssf = 0.0f;
    const size_t fbase = (size_t)n * CF_ * H_ * W_;
    #pragma unroll
    for (int j = 0; j < 3; ++j) {
        const int c = t + j * 256;
        const float* img = feats + fbase + (size_t)c * (H_ * W_);
        float v = img[i00] * w00 + img[i01] * w01 + img[i10] * w10 + img[i11] * w11;
        vf[j] = v; ssf += v * v;
    }
    float vc = 0.0f;
    if (t < CC_) {
        const float* img = code + ((size_t)n * CC_ + t) * (H_ * W_);
        vc = img[i00] * w00 + img[i01] * w01 + img[i10] * w10 + img[i11] * w11;
    }

    __shared__ float red[256];
    __shared__ float invF, invC;
    red[t] = ssf; __syncthreads();
    #pragma unroll
    for (int o = 128; o > 0; o >>= 1) { if (t < o) red[t] += red[t + o]; __syncthreads(); }
    if (t == 0) invF = 1.0f / fmaxf(sqrtf(red[0]), 1e-10f);
    __syncthreads();
    red[t] = (t < CC_) ? vc * vc : 0.0f; __syncthreads();
    #pragma unroll
    for (int o = 128; o > 0; o >>= 1) { if (t < o) red[t] += red[t + o]; __syncthreads(); }
    if (t == 0) invC = 1.0f / fmaxf(sqrtf(red[0]), 1e-10f);
    __syncthreads();

    _Float16* of = outF + ((size_t)n * PP_ + p) * CF_;
    #pragma unroll
    for (int j = 0; j < 3; ++j) of[t + j * 256] = (_Float16)(vf[j] * invF);
    if (t < CC_) outC[((size_t)n * PP_ + p) * CC_ + t] = (_Float16)(vc * invC);
}

// ---------------- kernel 2: WMMA correlation GEMM + fused loss reductions ----------------
// 256 threads = 8 waves; block tile 128x128 of the [1664 x 1664] correlation.
// Waves: 2 (p) x 4 (q); wave tile 64x32 -> 8 f32 accumulators, 4 A x 2 B frag reuse.
// Feature GEMM (K=768) streams operands through double-buffered LDS staged with
// async global->LDS copies (ASYNCcnt); fragments come from LDS (ds_load_b128).
__global__ __launch_bounds__(256) void k_corr(
    const _Float16* __restrict__ Af, const _Float16* __restrict__ Bf,
    const _Float16* __restrict__ Ac, const _Float16* __restrict__ Bc,
    float* __restrict__ rowsum_fd, float* __restrict__ rowsum_rc,
    float* __restrict__ scalars)
{
    __shared__ _Float16 smA[2][128 * LDST];
    __shared__ _Float16 smB[2][128 * LDST];

    const int n    = blockIdx.z;
    const int t    = threadIdx.x;
    const int wid  = t >> 5;
    const int lane = t & 31;
    const int g    = lane >> 4;              // half-wave
    const int m    = lane & 15;              // row (A) / column (B) within 16-tile
    const int wpi  = wid >> 2;               // 0..1  -> p offset wpi*64
    const int wqi  = wid & 3;                // 0..3  -> q offset wqi*32
    const int wp   = wpi * 64;
    const int wq   = wqi * 32;
    const int pbase = blockIdx.y * 128;
    const int qbase = blockIdx.x * 128;

    const _Float16* gA = Af + ((size_t)n * PP_ + pbase) * CF_;
    const _Float16* gB = Bf + ((size_t)n * PP_ + qbase) * CF_;

    // per-thread copy lanes: 2 chunks of A + 2 of B per K-step (512 chunks / tile)
    const int c0 = t, c1 = t + 256;
    const int r0 = c0 >> 2, s0 = c0 & 3;     // row, 8-half sub-chunk
    const int r1 = c1 >> 2, s1 = c1 & 3;

    auto stage = [&](int buf, int k) {
        cp_async16(&smA[buf][r0 * LDST + s0 * 8], gA + (size_t)r0 * CF_ + k + s0 * 8);
        cp_async16(&smB[buf][r0 * LDST + s0 * 8], gB + (size_t)r0 * CF_ + k + s0 * 8);
        cp_async16(&smA[buf][r1 * LDST + s1 * 8], gA + (size_t)r1 * CF_ + k + s1 * 8);
        cp_async16(&smB[buf][r1 * LDST + s1 * 8], gB + (size_t)r1 * CF_ + k + s1 * 8);
    };

    // ---- fd = Af x Bf^T, K = 768, double-buffered through LDS ----
    v8f accf[4][2] = {};
    stage(0, 0);
    for (int kt = 0; kt < KT_; ++kt) {
        const int cur = kt & 1;
        if (kt + 1 < KT_) {
            stage(cur ^ 1, (kt + 1) * 32);   // prefetch next tile (4 more async ops)
            cp_wait_le<4>();                 // current buffer's 4 copies done
        } else {
            cp_wait_le<0>();                 // last tile: drain everything
        }
        __syncthreads();

        const _Float16* As = smA[cur];
        const _Float16* Bs = smB[cur];
        v16h a[4], b[2];
        #pragma unroll
        for (int mi = 0; mi < 4; ++mi)
            a[mi] = load_a_frag(As + (wp + mi * 16 + m) * LDST + g * 8);
        #pragma unroll
        for (int ni = 0; ni < 2; ++ni)
            b[ni] = load_b_frag(Bs + (wq + ni * 16 + m) * LDST + g * 8);
        #pragma unroll
        for (int mi = 0; mi < 4; ++mi)
            #pragma unroll
            for (int ni = 0; ni < 2; ++ni)
                accf[mi][ni] = wmma_f16(a[mi], b[ni], accf[mi][ni]);

        __syncthreads();
    }

    // ---- cd = Ac x Bc^T, K = 64: tiny, direct from global ----
    v8f accc[4][2] = {};
    const _Float16* aC = Ac + ((size_t)n * PP_ + pbase + wp + m) * CC_ + g * 8;
    const _Float16* bC = Bc + ((size_t)n * PP_ + qbase + wq + m) * CC_ + g * 8;
    __builtin_prefetch(aC, 0, 3);
    __builtin_prefetch(bC, 0, 3);
    #pragma unroll
    for (int k = 0; k < CC_; k += 32) {
        v16h a[4], b[2];
        #pragma unroll
        for (int mi = 0; mi < 4; ++mi)
            a[mi] = load_a_frag(aC + (size_t)mi * 16 * CC_ + k);
        #pragma unroll
        for (int ni = 0; ni < 2; ++ni)
            b[ni] = load_b_frag(bC + (size_t)ni * 16 * CC_ + k);
        #pragma unroll
        for (int mi = 0; mi < 4; ++mi)
            #pragma unroll
            for (int ni = 0; ni < 2; ++ni)
                accc[mi][ni] = wmma_f16(a[mi], b[ni], accc[mi][ni]);
    }

    // ---- fused epilogue ----
    // C/D fragment layout: element (vgpr r, lane) -> row M = g*8 + r, col N = m.
    // Per-row sums over q: add the 2 ni-subtiles, reduce over the 16 column lanes,
    // one atomic per row from lane m==0 of each half-wave.
    #pragma unroll
    for (int mi = 0; mi < 4; ++mi) {
        #pragma unroll
        for (int r = 0; r < 8; ++r) {
            float rf = accf[mi][0][r] + accf[mi][1][r];
            float rc = fmaxf(accc[mi][0][r], 0.f) + fmaxf(accc[mi][1][r], 0.f);
            #pragma unroll
            for (int msk = 8; msk > 0; msk >>= 1) {
                rf += __shfl_xor(rf, msk, 16);
                rc += __shfl_xor(rc, msk, 16);
            }
            if (m == 0) {
                const size_t row = (size_t)n * PP_ + pbase + wp + mi * 16 + g * 8 + r;
                atomicAdd(&rowsum_fd[row], rf);
                atomicAdd(&rowsum_rc[row], rc);
            }
        }
    }

    // Global scalars: sum(fd), sum(relu(cd)), sum(relu(cd)*fd)
    float sfd = 0.f, src = 0.f, spr = 0.f;
    #pragma unroll
    for (int mi = 0; mi < 4; ++mi)
        #pragma unroll
        for (int ni = 0; ni < 2; ++ni)
            #pragma unroll
            for (int r = 0; r < 8; ++r) {
                float f = accf[mi][ni][r];
                float c = fmaxf(accc[mi][ni][r], 0.f);
                sfd += f; src += c; spr += c * f;
            }
    #pragma unroll
    for (int msk = 16; msk > 0; msk >>= 1) {
        sfd += __shfl_xor(sfd, msk, 32);
        src += __shfl_xor(src, msk, 32);
        spr += __shfl_xor(spr, msk, 32);
    }
    if (lane == 0) {
        atomicAdd(&scalars[0], sfd);
        atomicAdd(&scalars[1], src);
        atomicAdd(&scalars[2], spr);
    }
}

// ---------------- kernel 3: final combine ----------------
// loss = -( sum_prod - (1/S^2) * sum_p rowsum_fd[p]*rowsum_rc[p]
//           + (sum_fd / Mtot) * sum_rc ) / Mtot,  Mtot = N * S^4 (pads contribute 0)
__global__ __launch_bounds__(256) void k_final(
    const float* __restrict__ rowsum_fd, const float* __restrict__ rowsum_rc,
    const float* __restrict__ scalars, float* __restrict__ out)
{
    __shared__ float red[256];
    float acc = 0.0f;
    for (int i = threadIdx.x; i < N_ * PP_; i += 256)
        acc += rowsum_fd[i] * rowsum_rc[i];
    red[threadIdx.x] = acc; __syncthreads();
    #pragma unroll
    for (int o = 128; o > 0; o >>= 1) {
        if (threadIdx.x < o) red[threadIdx.x] += red[threadIdx.x + o];
        __syncthreads();
    }
    if (threadIdx.x == 0) {
        const float Mtot = (float)N_ * (float)P_ * (float)P_;  // 16 * 1600 * 1600
        const float sum_fd = scalars[0], sum_rc = scalars[1], sum_prod = scalars[2];
        const float T = red[0] / (float)P_;
        const float old_mean = sum_fd / Mtot;
        out[0] = -(sum_prod - T + old_mean * sum_rc) / Mtot;
    }
}

// ---------------- launch ----------------
extern "C" void kernel_launch(void* const* d_in, const int* in_sizes, int n_in,
                              void* d_out, int out_size, void* d_ws, size_t ws_size,
                              hipStream_t stream) {
    (void)in_sizes; (void)n_in; (void)out_size; (void)ws_size;
    const float* orig_feats     = (const float*)d_in[0];
    const float* orig_feats_pos = (const float*)d_in[1];
    const float* orig_code      = (const float*)d_in[2];
    const float* orig_code_pos  = (const float*)d_in[3];
    const float* coords1        = (const float*)d_in[4];
    const float* coords2        = (const float*)d_in[5];

    char* ws = (char*)d_ws;
    size_t off = 0;
    auto take = [&](size_t bytes) -> void* {
        void* p = ws + off;
        off += (bytes + 255) & ~(size_t)255;
        return p;
    };
    _Float16* Af = (_Float16*)take((size_t)N_ * PP_ * CF_ * sizeof(_Float16)); // 40.9 MB
    _Float16* Bf = (_Float16*)take((size_t)N_ * PP_ * CF_ * sizeof(_Float16)); // 40.9 MB
    _Float16* Ac = (_Float16*)take((size_t)N_ * PP_ * CC_ * sizeof(_Float16)); //  3.4 MB
    _Float16* Bc = (_Float16*)take((size_t)N_ * PP_ * CC_ * sizeof(_Float16)); //  3.4 MB
    float* rowsum_fd = (float*)take((size_t)N_ * PP_ * sizeof(float));         // 104 KB
    float* rowsum_rc = (float*)take((size_t)N_ * PP_ * sizeof(float));         // 104 KB
    float* scalars   = (float*)take(3 * sizeof(float));
    // rowsum_fd / rowsum_rc / scalars are contiguous (sizes are 256B multiples)
    const int nzero = N_ * PP_ * 2 + 3;

    k_zero<<<(nzero + 255) / 256, 256, 0, stream>>>(rowsum_fd, nzero);
    const int padF16 = N_ * (PP_ - P_) * CF_ / 8;   // 8-half chunks in feat pads
    const int padC16 = N_ * (PP_ - P_) * CC_ / 8;
    k_zero_pad<<<(padF16 + 255) / 256, 256, 0, stream>>>(Af, CF_, padF16);
    k_zero_pad<<<(padF16 + 255) / 256, 256, 0, stream>>>(Bf, CF_, padF16);
    k_zero_pad<<<(padC16 + 255) / 256, 256, 0, stream>>>(Ac, CC_, padC16);
    k_zero_pad<<<(padC16 + 255) / 256, 256, 0, stream>>>(Bc, CC_, padC16);

    k_sample<<<dim3(P_, N_), 256, 0, stream>>>(orig_feats,     orig_code,     coords1, Af, Ac);
    k_sample<<<dim3(P_, N_), 256, 0, stream>>>(orig_feats_pos, orig_code_pos, coords2, Bf, Bc);
    k_corr<<<dim3(PP_ / 128, PP_ / 128, N_), 256, 0, stream>>>(Af, Bf, Ac, Bc,
                                                               rowsum_fd, rowsum_rc, scalars);
    k_final<<<1, 256, 0, stream>>>(rowsum_fd, rowsum_rc, scalars, (float*)d_out);
}
